// EfficientMoE_64922725646673
// MI455X (gfx1250) — matrix operand order repaired and checked by reading, not compiled, not run on hardware
//
#include <hip/hip_runtime.h>
#include <math.h>

#define T_TOK 32768
#define DIMK  512
#define HID   2048
#define NEXP  8
#define CAP   4915
#define NB    4096
#define LDA   40            // LDS row stride in halves (64B data + 16B pad, 16B-aligned)

typedef __attribute__((ext_vector_type(16))) _Float16 v16h;
typedef __attribute__((ext_vector_type(8)))  _Float16 v8h;
typedef __attribute__((ext_vector_type(8)))  float    v8f;
typedef __attribute__((ext_vector_type(4)))  unsigned int u32x4;
typedef __attribute__((ext_vector_type(8)))  int i32x8;
typedef __attribute__((ext_vector_type(4)))  int i32x4;

// ---------------------------------------------------------------------------
// helpers
// ---------------------------------------------------------------------------
__device__ __forceinline__ v16h make_frag(const _Float16* p0, const _Float16* p1) {
    v8h lo = *(const v8h*)p0;
    v8h hi = *(const v8h*)p1;
    v16h r;
#pragma unroll
    for (int i = 0; i < 8; ++i) { r[i] = lo[i]; r[i + 8] = hi[i]; }
    return r;
}

__device__ __forceinline__ float gelu_exact(float v) {
    return 0.5f * v * (1.0f + erff(v * 0.70710678118654752f));
}

__device__ __forceinline__ int bucket_of(float w) {
    int b = (int)(w * (float)NB);
    if (b < 0) b = 0;
    if (b > NB - 1) b = NB - 1;
    return b;
}

// ---------------------------------------------------------------------------
// Tensor Data Mover: 2D tile (tile_d0 x tile_d1 elements of 2 bytes) from
// row-major global (row stride = stride_elems) into LDS with 16B row pad
// (pad_interval code 3 = every 16 DWORDs, pad_amount code 3 = 4 DWORDs),
// matching the LDA=40-half skewed LDS layout. D# layout per CDNA5 ISA ch.8.
// This toolchain exposes the 6-arg builtin form (extra i32x8 group, zeroed).
// ---------------------------------------------------------------------------
__device__ __forceinline__ void tdm_load_2d_f16(unsigned int lds_byte_addr,
                                                const void* gaddr,
                                                unsigned int tile_d0,
                                                unsigned int tile_d1,
                                                unsigned int tensor_d0,
                                                unsigned int tensor_d1,
                                                unsigned int stride_elems) {
    unsigned long long ga = (unsigned long long)(uintptr_t)gaddr;
    u32x4 g0;
    g0[0] = 1u;                                            // count=1, user D#
    g0[1] = lds_byte_addr;                                 // lds_addr
    g0[2] = (unsigned int)(ga & 0xffffffffu);              // global_addr[31:0]
    g0[3] = (unsigned int)((ga >> 32) & 0x01ffffffu)       // global_addr[56:32]
          | (2u << 30);                                    // type = 2 ("image")
    i32x8 g1;
    g1[0] = (int)((1u << 16)      // data_size = 1 -> 2 bytes
                | (1u << 20)      // pad_enable
                | (3u << 22)      // pad_interval: 16 DWORDs (64B data row)
                | (3u << 25));    // pad_amount: 4 DWORDs (16B skew)
    g1[1] = (int)((tensor_d0 & 0xffffu) << 16);                          // t_dim0[15:0]
    g1[2] = (int)(((tensor_d0 >> 16) & 0xffffu) | ((tensor_d1 & 0xffffu) << 16));
    g1[3] = (int)(((tensor_d1 >> 16) & 0xffffu) | ((tile_d0 & 0xffffu) << 16));
    g1[4] = (int)(tile_d1 & 0xffffu);                                    // tile_dim2=0
    g1[5] = (int)stride_elems;                                           // d0_stride lo
    g1[6] = 0;                                                           // d0_stride hi
    g1[7] = 0;                                                           // d1_stride (2D)
    i32x4 g2 = {1, 0, 0, 0};   // tensor_dim2=1; tile_dim3=0
    i32x4 g3 = {0, 0, 0, 0};
    i32x8 g4 = {0, 0, 0, 0, 0, 0, 0, 0};   // extra group (6-arg form), unused
    __builtin_amdgcn_tensor_load_to_lds(g0, g1, g2, g3, g4, 0);
}

// ---------------------------------------------------------------------------
// precision conversion
// ---------------------------------------------------------------------------
__global__ __launch_bounds__(256) void cvt_f32_f16(const float* __restrict__ src,
                                                   _Float16* __restrict__ dst, int n) {
    int i = blockIdx.x * 256 + threadIdx.x;
    if (i < n) dst[i] = (_Float16)src[i];
}

// src: [NEXP][R][C] row-major f32 -> dst: [NEXP][C][R] f16 (transposed per expert)
__global__ __launch_bounds__(256) void cvt_transpose(const float* __restrict__ src,
                                                     _Float16* __restrict__ dst,
                                                     int R, int C) {
    long i = (long)blockIdx.x * 256 + threadIdx.x;   // output-linear: ((e*C + c)*R + r)
    long total = (long)NEXP * R * C;
    if (i >= total) return;
    int r = (int)(i % R);
    long t = i / R;
    int c = (int)(t % C);
    int e = (int)(t / C);
    dst[i] = (_Float16)src[((long)e * R + r) * C + c];
}

// ---------------------------------------------------------------------------
// gating: logits -> softmax -> top-2
// ---------------------------------------------------------------------------
__global__ __launch_bounds__(256) void gate_kernel(const float* __restrict__ x,
                                                   const float* __restrict__ gw,
                                                   const float* __restrict__ gb,
                                                   int* __restrict__ topi,
                                                   float* __restrict__ topw) {
    int t = blockIdx.x * 256 + threadIdx.x;
    if (t >= T_TOK) return;
    const float4* xr = (const float4*)(x + (size_t)t * DIMK);

    float lg[NEXP];
#pragma unroll
    for (int e = 0; e < NEXP; ++e) {
        const float4* wr = (const float4*)(gw + e * DIMK);
        float s = 0.f;
        for (int i = 0; i < DIMK / 4; ++i) {
            float4 a = xr[i], b = wr[i];
            s += a.x * b.x + a.y * b.y + a.z * b.z + a.w * b.w;
        }
        lg[e] = s + gb[e];
    }
    float mx = lg[0];
#pragma unroll
    for (int e = 1; e < NEXP; ++e) mx = fmaxf(mx, lg[e]);
    float p[NEXP], den = 0.f;
#pragma unroll
    for (int e = 0; e < NEXP; ++e) { p[e] = __expf(lg[e] - mx); den += p[e]; }
    float inv = 1.f / den;
#pragma unroll
    for (int e = 0; e < NEXP; ++e) p[e] *= inv;

    int i0 = 0; float w0 = p[0];
#pragma unroll
    for (int e = 1; e < NEXP; ++e) if (p[e] > w0) { w0 = p[e]; i0 = e; }
    int i1 = -1; float w1 = -1.f;
#pragma unroll
    for (int e = 0; e < NEXP; ++e)
        if (e != i0 && p[e] > w1) { w1 = p[e]; i1 = e; }

    topi[2 * t] = i0; topi[2 * t + 1] = i1;
    topw[2 * t] = w0; topw[2 * t + 1] = w1;
}

// ---------------------------------------------------------------------------
// capacity selection: histogram -> threshold -> deterministic pick
// ---------------------------------------------------------------------------
__global__ __launch_bounds__(256) void hist_kernel(const int* __restrict__ topi,
                                                   const float* __restrict__ topw,
                                                   int* __restrict__ hist) {
    int t = blockIdx.x * 256 + threadIdx.x;
    if (t >= T_TOK) return;
#pragma unroll
    for (int k = 0; k < 2; ++k) {
        int e = topi[2 * t + k];
        float w = topw[2 * t + k];
        atomicAdd(&hist[e * NB + bucket_of(w)], 1);
    }
}

__global__ __launch_bounds__(32) void thresh_kernel(const int* __restrict__ hist,
                                                    int* __restrict__ thresh,
                                                    int* __restrict__ remeq) {
    int e = threadIdx.x;
    if (e >= NEXP) return;
    int cum = 0, th = -1, req = 0;
    for (int b = NB - 1; b >= 0; --b) {
        int c = hist[e * NB + b];
        if (cum + c >= CAP) { th = b; req = CAP - cum; break; }
        cum += c;
    }
    thresh[e] = th; remeq[e] = req;   // th == -1 => fewer routed than cap: take all
}

__global__ __launch_bounds__(256) void select_main(const int* __restrict__ topi,
                                                   const float* __restrict__ topw,
                                                   const int* __restrict__ thresh,
                                                   int* __restrict__ nsel,
                                                   int* __restrict__ ncand,
                                                   int* __restrict__ sel_idx,
                                                   float* __restrict__ sel_w,
                                                   int* __restrict__ cand_t,
                                                   float* __restrict__ cand_w) {
    int t = blockIdx.x * 256 + threadIdx.x;
    if (t >= T_TOK) return;
#pragma unroll
    for (int k = 0; k < 2; ++k) {
        int e = topi[2 * t + k];
        float w = topw[2 * t + k];
        int b = bucket_of(w);
        int th = thresh[e];
        if (b > th) {
            int s = atomicAdd(&nsel[e], 1);
            if (s < CAP) { sel_idx[e * CAP + s] = t; sel_w[e * CAP + s] = w; }
        } else if (b == th) {
            int ci = atomicAdd(&ncand[e], 1);
            cand_t[(size_t)e * T_TOK + ci] = t;
            cand_w[(size_t)e * T_TOK + ci] = w;
        }
    }
}

// exact (weight, index) rank inside the threshold bucket -> deterministic set
__global__ __launch_bounds__(256) void select_eq(const int* __restrict__ ncand,
                                                 const int* __restrict__ remeq,
                                                 const int* __restrict__ cand_t,
                                                 const float* __restrict__ cand_w,
                                                 int* __restrict__ nsel,
                                                 int* __restrict__ sel_idx,
                                                 float* __restrict__ sel_w) {
    int e = blockIdx.y;
    int i = blockIdx.x * 256 + threadIdx.x;
    int nc = ncand[e];
    if (i >= nc) return;
    int need = remeq[e];
    if (need <= 0) return;
    float wi = cand_w[(size_t)e * T_TOK + i];
    int   ti = cand_t[(size_t)e * T_TOK + i];
    int rank = 0;
    for (int j = 0; j < nc; ++j) {
        float wj = cand_w[(size_t)e * T_TOK + j];
        int   tj = cand_t[(size_t)e * T_TOK + j];
        if (wj > wi || (wj == wi && tj < ti)) ++rank;
    }
    if (rank < need) {
        int s = atomicAdd(&nsel[e], 1);
        if (s < CAP) { sel_idx[e * CAP + s] = ti; sel_w[e * CAP + s] = wi; }
    }
}

// ---------------------------------------------------------------------------
// FFN1: h = gelu(x[sel] @ w1 + b1)
// A: gathered token rows (vector-staged, indexed), B: w1^T tile via TDM.
// block: 256 thr (8 waves), tile 128(M)x128(N), K-step 32
// ---------------------------------------------------------------------------
__global__ __launch_bounds__(256) void ffn1_kernel(const _Float16* __restrict__ xh,
                                                   const int* __restrict__ seli_e,
                                                   const _Float16* __restrict__ w1t_e,
                                                   const float* __restrict__ b1_e,
                                                   _Float16* __restrict__ hbuf) {
    __shared__ __align__(16) _Float16 As[128 * LDA];
    __shared__ __align__(16) _Float16 Bs[128 * LDA];

    const int m0 = blockIdx.x * 128, n0 = blockIdx.y * 128;
    const int tid = threadIdx.x;
    const int wave = tid >> 5, lane = tid & 31;
    const int wr = wave & 1, wc = wave >> 1;
    const int half = lane >> 4, ln = lane & 15;

    const int rloc = tid & 127, kc = tid >> 7;        // A staging: 2 thr/row, 16 halves
    int mrow = m0 + rloc;
    int tok = (mrow < CAP) ? seli_e[mrow] : 0;
    const _Float16* arow = xh + (size_t)tok * DIMK;
    const unsigned int bs_lds = (unsigned int)(uintptr_t)&Bs[0];

    v8f acc[4][2];
#pragma unroll
    for (int i = 0; i < 4; ++i)
#pragma unroll
        for (int j = 0; j < 2; ++j) acc[i][j] = (v8f)(0.f);

    for (int kk = 0; kk < DIMK; kk += 32) {
        // A tile: indexed gather, vector-staged
        const uint4* ap = (const uint4*)(arow + kk + kc * 16);
        uint4 a0 = ap[0], a1 = ap[1];
        *(uint4*)&As[rloc * LDA + kc * 16]     = a0;
        *(uint4*)&As[rloc * LDA + kc * 16 + 8] = a1;
        if (kk + 32 < DIMK) __builtin_prefetch(arow + kk + 32 + kc * 16, 0, 0);

        // B tile: Tensor Data Mover DMA (wave 0 issues for the workgroup)
        if (wave == 0) {
            tdm_load_2d_f16(bs_lds, w1t_e + (size_t)n0 * DIMK + kk,
                            /*tile*/ 32, 128, /*tensor*/ DIMK, 128, DIMK);
            __builtin_amdgcn_s_wait_tensorcnt(0);
        }
        __syncthreads();

        v16h bf[2];
#pragma unroll
        for (int j = 0; j < 2; ++j) {
            const _Float16* p = &Bs[(wc * 32 + j * 16 + ln) * LDA + half * 8];
            bf[j] = make_frag(p, p + 16);
        }
#pragma unroll
        for (int i = 0; i < 4; ++i) {
            const _Float16* p = &As[(wr * 64 + i * 16 + ln) * LDA];
            v16h af = make_frag(p + half * 8, p + 16 + half * 8);
#pragma unroll
            for (int j = 0; j < 2; ++j)
                acc[i][j] = __builtin_amdgcn_wmma_f32_16x16x32_f16(
                    false, af, false, bf[j], (short)0, acc[i][j], false, false);
        }
        __syncthreads();
    }

#pragma unroll
    for (int i = 0; i < 4; ++i)
#pragma unroll
        for (int j = 0; j < 2; ++j) {
            int n = n0 + wc * 32 + j * 16 + ln;
            float bias = b1_e[n];
#pragma unroll
            for (int r = 0; r < 8; ++r) {
                int m = m0 + wr * 64 + i * 16 + half * 8 + r;
                if (m < CAP)
                    hbuf[(size_t)m * HID + n] = (_Float16)gelu_exact(acc[i][j][r] + bias);
            }
        }
}

// ---------------------------------------------------------------------------
// FFN2: out[sel_idx] += (h @ w2 + b2) * sel_w   (K = 2048)
// Both A (h, dense) and B (w2^T) tiles staged by TDM; rows past cap zero-fill
// via tensor_dim1 OOB.
// ---------------------------------------------------------------------------
__global__ __launch_bounds__(256) void ffn2_kernel(const _Float16* __restrict__ hbuf,
                                                   const _Float16* __restrict__ w2t_e,
                                                   const float* __restrict__ b2_e,
                                                   const int* __restrict__ seli_e,
                                                   const float* __restrict__ selw_e,
                                                   float* __restrict__ out) {
    __shared__ __align__(16) _Float16 As[128 * LDA];
    __shared__ __align__(16) _Float16 Bs[128 * LDA];

    const int m0 = blockIdx.x * 128, n0 = blockIdx.y * 128;
    const int tid = threadIdx.x;
    const int wave = tid >> 5, lane = tid & 31;
    const int wr = wave & 1, wc = wave >> 1;
    const int half = lane >> 4, ln = lane & 15;

    const unsigned int as_lds = (unsigned int)(uintptr_t)&As[0];
    const unsigned int bs_lds = (unsigned int)(uintptr_t)&Bs[0];
    const unsigned int arows = (unsigned int)(CAP - m0);   // rows in-tensor (zero-fill past)

    v8f acc[4][2];
#pragma unroll
    for (int i = 0; i < 4; ++i)
#pragma unroll
        for (int j = 0; j < 2; ++j) acc[i][j] = (v8f)(0.f);

    for (int kk = 0; kk < HID; kk += 32) {
        if (wave == 0) {
            tdm_load_2d_f16(as_lds, hbuf + (size_t)m0 * HID + kk,
                            /*tile*/ 32, 128, /*tensor*/ HID, arows, HID);
            __builtin_amdgcn_s_wait_tensorcnt(0);
        } else if (wave == 1) {
            tdm_load_2d_f16(bs_lds, w2t_e + (size_t)n0 * HID + kk,
                            /*tile*/ 32, 128, /*tensor*/ HID, 128, HID);
            __builtin_amdgcn_s_wait_tensorcnt(0);
        }
        __syncthreads();

        v16h bf[2];
#pragma unroll
        for (int j = 0; j < 2; ++j) {
            const _Float16* p = &Bs[(wc * 32 + j * 16 + ln) * LDA + half * 8];
            bf[j] = make_frag(p, p + 16);
        }
#pragma unroll
        for (int i = 0; i < 4; ++i) {
            const _Float16* p = &As[(wr * 64 + i * 16 + ln) * LDA];
            v16h af = make_frag(p + half * 8, p + 16 + half * 8);
#pragma unroll
            for (int j = 0; j < 2; ++j)
                acc[i][j] = __builtin_amdgcn_wmma_f32_16x16x32_f16(
                    false, af, false, bf[j], (short)0, acc[i][j], false, false);
        }
        __syncthreads();
    }

#pragma unroll
    for (int i = 0; i < 4; ++i)
#pragma unroll
        for (int j = 0; j < 2; ++j) {
            int n = n0 + wc * 32 + j * 16 + ln;
            float bias = b2_e[n];
#pragma unroll
            for (int r = 0; r < 8; ++r) {
                int m = m0 + wr * 64 + i * 16 + half * 8 + r;
                if (m < CAP) {
                    float w = selw_e[m];
                    if (w > 0.f) {
                        int tok = seli_e[m];
                        out[(size_t)tok * DIMK + n] += (acc[i][j][r] + bias) * w;
                    }
                }
            }
        }
}

// ---------------------------------------------------------------------------
// launch
// ---------------------------------------------------------------------------
extern "C" void kernel_launch(void* const* d_in, const int* in_sizes, int n_in,
                              void* d_out, int out_size, void* d_ws, size_t ws_size,
                              hipStream_t stream) {
    (void)in_sizes; (void)n_in; (void)out_size; (void)ws_size;
    const float* x  = (const float*)d_in[0];
    const float* gw = (const float*)d_in[1];
    const float* gb = (const float*)d_in[2];
    const float* w1 = (const float*)d_in[3];
    const float* b1 = (const float*)d_in[4];
    const float* w2 = (const float*)d_in[5];
    const float* b2 = (const float*)d_in[6];
    float* out = (float*)d_out;

    char* ws = (char*)d_ws;
    size_t off = 0;
    auto alloc = [&](size_t bytes) -> char* {
        char* p = ws + off;
        off += (bytes + 255) & ~(size_t)255;
        return p;
    };
    _Float16* xh    = (_Float16*)alloc((size_t)T_TOK * DIMK * 2);
    _Float16* w1t   = (_Float16*)alloc((size_t)NEXP * HID * DIMK * 2);
    _Float16* w2t   = (_Float16*)alloc((size_t)NEXP * DIMK * HID * 2);
    _Float16* hbuf  = (_Float16*)alloc((size_t)CAP * HID * 2);
    int*      topi  = (int*)alloc((size_t)T_TOK * 2 * 4);
    float*    topw  = (float*)alloc((size_t)T_TOK * 2 * 4);
    int*      hist  = (int*)alloc((size_t)NEXP * NB * 4);
    int*      thr   = (int*)alloc(NEXP * 4);
    int*      remeq = (int*)alloc(NEXP * 4);
    int*      nsel  = (int*)alloc(NEXP * 4);
    int*      ncand = (int*)alloc(NEXP * 4);
    int*      seli  = (int*)alloc((size_t)NEXP * CAP * 4);
    float*    selw  = (float*)alloc((size_t)NEXP * CAP * 4);
    int*      candt = (int*)alloc((size_t)NEXP * T_TOK * 4);
    float*    candw = (float*)alloc((size_t)NEXP * T_TOK * 4);

    (void)hipMemsetAsync(out, 0, (size_t)T_TOK * DIMK * 4, stream);
    (void)hipMemsetAsync(hist, 0, (size_t)NEXP * NB * 4, stream);
    (void)hipMemsetAsync(nsel, 0, NEXP * 4, stream);
    (void)hipMemsetAsync(ncand, 0, NEXP * 4, stream);
    (void)hipMemsetAsync(seli, 0, (size_t)NEXP * CAP * 4, stream);
    (void)hipMemsetAsync(selw, 0, (size_t)NEXP * CAP * 4, stream);

    // precision conversion (+ weight transposes so GEMM B is contiguous in K)
    {
        int n = T_TOK * DIMK;
        cvt_f32_f16<<<(n + 255) / 256, 256, 0, stream>>>(x, xh, n);
        long t1 = (long)NEXP * DIMK * HID;
        cvt_transpose<<<(unsigned)((t1 + 255) / 256), 256, 0, stream>>>(w1, w1t, DIMK, HID);
        long t2 = (long)NEXP * HID * DIMK;
        cvt_transpose<<<(unsigned)((t2 + 255) / 256), 256, 0, stream>>>(w2, w2t, HID, DIMK);
    }

    // gating + capacity-limited selection
    gate_kernel<<<T_TOK / 256, 256, 0, stream>>>(x, gw, gb, topi, topw);
    hist_kernel<<<T_TOK / 256, 256, 0, stream>>>(topi, topw, hist);
    thresh_kernel<<<1, 32, 0, stream>>>(hist, thr, remeq);
    select_main<<<T_TOK / 256, 256, 0, stream>>>(topi, topw, thr, nsel, ncand,
                                                 seli, selw, candt, candw);
    select_eq<<<dim3(T_TOK / 256, NEXP), 256, 0, stream>>>(ncand, remeq, candt, candw,
                                                           nsel, seli, selw);

    // per-expert FFN (serialized on stream -> scatter += needs no atomics)
    const int MB = (CAP + 127) / 128;   // 39
    for (int e = 0; e < NEXP; ++e) {
        ffn1_kernel<<<dim3(MB, HID / 128), 256, 0, stream>>>(
            xh, seli + e * CAP, w1t + (size_t)e * HID * DIMK, b1 + e * HID, hbuf);
        ffn2_kernel<<<dim3(MB, DIMK / 128), 256, 0, stream>>>(
            hbuf, w2t + (size_t)e * DIMK * HID, b2 + e * DIMK,
            seli + e * CAP, selw + e * CAP, out);
    }
}
